// CrossViewDeformableBlock_84653805404295
// MI455X (gfx1250) — matrix operand order repaired and checked by reading, not compiled, not run on hardware
//
#include <hip/hip_runtime.h>
#include <hip/hip_bf16.h>
#include <stdint.h>

// ---------------- problem constants ----------------
#define B_     2
#define NCAM   6
#define DIM_   128
#define HW_    4096          // 64*64
#define HIMG   32
#define WIMG   88
#define HWIMG  2816          // 32*88
#define HEADS_ 4
#define DHEAD  32
#define INNER_ 128
#define NPTS   8

typedef __attribute__((ext_vector_type(16))) _Float16 v16h;
typedef __attribute__((ext_vector_type(8)))  _Float16 v8h;
typedef __attribute__((ext_vector_type(8)))  float    v8f;

// =====================================================================
// Kernel 1: camera projection -> normalized grid[bn][hw][2]
// =====================================================================
__global__ void cvd_grid_kernel(const float* __restrict__ Kmat,
                                const float* __restrict__ Emat,
                                const float* __restrict__ wxy,
                                float* __restrict__ grid) {
    int idx = blockIdx.x * blockDim.x + threadIdx.x;
    if (idx >= B_ * NCAM * HW_) return;
    int hw = idx & (HW_ - 1);
    int bn = idx >> 12;

    float wx = wxy[hw];
    float wy = wxy[HW_ + hw];

    const float* E = Emat + bn * 16;
    float c0 = E[0] * wx + E[1] * wy + E[3];
    float c1 = E[4] * wx + E[5] * wy + E[7];
    float c2 = E[8] * wx + E[9] * wy + E[11];

    const float* K = Kmat + bn * 9;
    float px = K[0] * c0 + K[1] * c1 + K[2] * c2;
    float py = K[3] * c0 + K[4] * c1 + K[5] * c2;
    float pz = K[6] * c0 + K[7] * c1 + K[8] * c2;
    float dz = fmaxf(pz, 1e-6f);

    grid[(size_t)idx * 2 + 0] = (px / dz) / (float)(WIMG - 1) * 2.0f - 1.0f;
    grid[(size_t)idx * 2 + 1] = (py / dz) / (float)(HIMG - 1) * 2.0f - 1.0f;
}

// =====================================================================
// Kernel 2a: f32 -> f16 flat copy (weights)
// =====================================================================
__global__ void cvd_cvt_f16_kernel(const float* __restrict__ src,
                                   _Float16* __restrict__ dst, int n) {
    int i = blockIdx.x * blockDim.x + threadIdx.x;
    if (i < n) dst[i] = (_Float16)src[i];
}

// =====================================================================
// Kernel 2b: f32 [R,C] -> f16 [C,R] LDS-tiled transpose (batched via z)
// =====================================================================
__global__ void cvd_cvt_transpose_kernel(const float* __restrict__ src,
                                         _Float16* __restrict__ dst,
                                         int R, int C) {
    __shared__ _Float16 tile[16][17];
    int c0 = blockIdx.x << 4, r0 = blockIdx.y << 4;
    const float* s = src + (size_t)blockIdx.z * R * C;
    _Float16*    d = dst + (size_t)blockIdx.z * R * C;
    int tx = threadIdx.x, ty = threadIdx.y;
    tile[ty][tx] = (_Float16)s[(size_t)(r0 + ty) * C + c0 + tx];
    __syncthreads();
    d[(size_t)(c0 + ty) * R + r0 + tx] = tile[tx][ty];
}

// =====================================================================
// Kernel 3: batched WMMA GEMM, K fixed = 128.
//   A : f16 [M,128]  row-major (weights, shared over batch)
//   Xt: f16 [N,128]  row-major (= X transposed), batch stride xtStride
//   mode 0: Y = f32 [M,N]  (+bias, optional relu)
//   mode 1: Y = f16 [N,M]  (transposed, packed v8h stores)
// One wave = one 16x16 tile, 4 unrolled v_wmma_f32_16x16x32_f16.
// =====================================================================
__global__ void cvd_wmma_gemm_kernel(const _Float16* __restrict__ A,
                                     const _Float16* __restrict__ Xt,
                                     const float* __restrict__ bias,
                                     void* __restrict__ Y,
                                     int M, int N, int batch,
                                     long xtStride, long yStride,
                                     int relu, int mode) {
    const int lane = threadIdx.x & 31;
    const int half = lane >> 4;
    const int lr   = lane & 15;

    const int wavesPerBlock = blockDim.x >> 5;
    const int waveId   = blockIdx.x * wavesPerBlock + (threadIdx.x >> 5);
    const int numWaves = gridDim.x * wavesPerBlock;

    const int mt = M >> 4;
    const int nt = N >> 4;
    const long tilesPerBatch = (long)mt * nt;
    const long totalTiles = tilesPerBatch * batch;

    for (long t = waveId; t < totalTiles; t += numWaves) {
        int bIdx = (int)(t / tilesPerBatch);
        int rem  = (int)(t - (long)bIdx * tilesPerBatch);
        int tm = rem / nt;
        int tn = rem - tm * nt;
        int mBase = tm << 4;
        int nBase = tn << 4;

        const _Float16* Ar = A + (size_t)(mBase + lr) * 128;
        const _Float16* Br = Xt + (size_t)bIdx * xtStride + (size_t)(nBase + lr) * 128;

        v8f c = {};
        #pragma unroll
        for (int kk = 0; kk < 128; kk += 32) {
            // A frag: lane<16 -> K {kk..kk+7, kk+16..kk+23}; lane>=16 -> +8
            v8h a0 = *(const v8h*)(Ar + kk + half * 8);
            v8h a1 = *(const v8h*)(Ar + kk + 16 + half * 8);
            v16h a = __builtin_shufflevector(a0, a1,
                        0,1,2,3,4,5,6,7,8,9,10,11,12,13,14,15);
            // B frag: lane<16 -> K {kk..kk+7, kk+16..kk+23}; lane>=16 -> +8
            v8h b0 = *(const v8h*)(Br + kk + half * 8);
            v8h b1 = *(const v8h*)(Br + kk + 16 + half * 8);
            v16h b = __builtin_shufflevector(b0, b1,
                        0,1,2,3,4,5,6,7,8,9,10,11,12,13,14,15);
            c = __builtin_amdgcn_wmma_f32_16x16x32_f16(
                    false, a, false, b, (short)0, c, false, false);
        }

        int col = nBase + lr;
        if (mode == 1) {
            // f16 transposed output: lane's 8 rows are contiguous -> one b128
            v8h pk;
            #pragma unroll
            for (int i = 0; i < 8; ++i) {
                float v = c[i] + bias[mBase + half * 8 + i];
                if (relu) v = fmaxf(v, 0.0f);
                pk[i] = (_Float16)v;
            }
            *(v8h*)((_Float16*)Y + (size_t)bIdx * yStride +
                    (size_t)col * M + mBase + half * 8) = pk;
        } else {
            #pragma unroll
            for (int i = 0; i < 8; ++i) {
                int row = mBase + half * 8 + i;
                float v = c[i] + bias[row];
                if (relu) v = fmaxf(v, 0.0f);
                ((float*)Y)[(size_t)bIdx * yStride + (size_t)row * N + col] = v;
            }
        }
    }
}

// =====================================================================
// Kernel 4: fused deformable sampling + per-point softmax attention.
// One wave per (b, hw). lane = head*8 + point.
//   qT  : f16 [b][hw][128]        (async-staged to LDS, one b64/lane)
//   kvT : f16 [bn][y*w+x][256]    (spatial-major: 4 taps x wide b128 loads)
//   out : wgtT f16 [b][hw][128]   (packed v8h stores)
// =====================================================================
__global__ void cvd_sample_attn_kernel(const float* __restrict__ grid,
                                       const float* __restrict__ off,
                                       const _Float16* __restrict__ qT,
                                       const _Float16* __restrict__ kvT,
                                       _Float16* __restrict__ wgtT) {
    __shared__ _Float16 smem_q[8 * INNER_];   // 8 waves/block * 128 halves

    const int lane = threadIdx.x & 31;
    const int warp = threadIdx.x >> 5;
    const int idx  = blockIdx.x * 8 + warp;   // 0 .. B_*HW_-1
    const int bb = idx >> 12;
    const int hw = idx & (HW_ - 1);
    const int head = lane >> 3;
    const int p    = lane & 7;

    // ---- async-stage qT row (256B) into LDS: one b64 per lane ----
    _Float16* qs = &smem_q[warp * INNER_];
    {
        unsigned ldsBase = (unsigned)(uintptr_t)qs;
        const _Float16* gq = qT + ((size_t)bb * HW_ + hw) * INNER_;
        unsigned laddr = ldsBase + (unsigned)(lane * 8);
        const _Float16* ga = gq + lane * 4;
        asm volatile("global_load_async_to_lds_b64 %0, %1, off"
                     :: "v"(laddr), "v"(ga) : "memory");
        asm volatile("s_wait_asynccnt 0" ::: "memory");
    }
    const _Float16* qh = qs + head * DHEAD;

    float offx = off[((size_t)bb * (2 * NPTS) + 2 * p + 0) * HW_ + hw];
    float offy = off[((size_t)bb * (2 * NPTS) + 2 * p + 1) * HW_ + hw];

    float vacc[DHEAD];
    #pragma unroll
    for (int d = 0; d < DHEAD; ++d) vacc[d] = 0.0f;

    for (int n = 0; n < NCAM; ++n) {
        const float* g = grid + (((size_t)(bb * NCAM + n) * HW_) + hw) * 2;
        float sx = fminf(fmaxf(g[0] + offx, -1.0f), 1.0f);
        float sy = fminf(fmaxf(g[1] + offy, -1.0f), 1.0f);
        float ix = (sx + 1.0f) * 0.5f * (float)(WIMG - 1);
        float iy = (sy + 1.0f) * 0.5f * (float)(HIMG - 1);
        float x0f = floorf(ix), y0f = floorf(iy);
        float fx = ix - x0f,    fy = iy - y0f;
        int x0 = min(max((int)x0f, 0), WIMG - 1);
        int x1 = min(max((int)x0f + 1, 0), WIMG - 1);
        int y0 = min(max((int)y0f, 0), HIMG - 1);
        int y1 = min(max((int)y0f + 1, 0), HIMG - 1);
        int   taps[4] = { y0 * WIMG + x0, y0 * WIMG + x1,
                          y1 * WIMG + x0, y1 * WIMG + x1 };
        float wts[4]  = { (1.0f - fx) * (1.0f - fy), fx * (1.0f - fy),
                          (1.0f - fx) * fy,          fx * fy };

        const _Float16* kvb = kvT + (size_t)(bb * NCAM + n) * HWIMG * (2 * INNER_);

        float ks[DHEAD], vs[DHEAD];
        #pragma unroll
        for (int d = 0; d < DHEAD; ++d) { ks[d] = 0.0f; vs[d] = 0.0f; }

        #pragma unroll
        for (int t = 0; t < 4; ++t) {
            const _Float16* base = kvb + (size_t)taps[t] * (2 * INNER_) + head * DHEAD;
            __builtin_prefetch(base, 0, 1);
            const v8h* kp = (const v8h*)base;
            const v8h* vp = (const v8h*)(base + INNER_);
            #pragma unroll
            for (int j = 0; j < 4; ++j) {
                v8h kj = kp[j];
                v8h vj = vp[j];
                #pragma unroll
                for (int i = 0; i < 8; ++i) {
                    ks[j * 8 + i] += wts[t] * (float)kj[i];
                    vs[j * 8 + i] += wts[t] * (float)vj[i];
                }
            }
        }

        float sim = 0.0f;
        #pragma unroll
        for (int d = 0; d < DHEAD; ++d) sim += (float)qh[d] * ks[d];

        // softmax over the 8 point-lanes of this head group
        float m = sim;
        m = fmaxf(m, __shfl_xor(m, 1, 8));
        m = fmaxf(m, __shfl_xor(m, 2, 8));
        m = fmaxf(m, __shfl_xor(m, 4, 8));
        float e = __expf(sim - m);
        float s = e;
        s += __shfl_xor(s, 1, 8);
        s += __shfl_xor(s, 2, 8);
        s += __shfl_xor(s, 4, 8);
        float att = e / s;

        #pragma unroll
        for (int d = 0; d < DHEAD; ++d) vacc[d] += att * vs[d];
    }

    // mean over cameras, then reduce over the 8 point-lanes
    const float invN = 1.0f / (float)NCAM;
    #pragma unroll
    for (int d = 0; d < DHEAD; ++d) {
        float v = vacc[d] * invN;
        v += __shfl_xor(v, 1, 8);
        v += __shfl_xor(v, 2, 8);
        v += __shfl_xor(v, 4, 8);
        vacc[d] = v;
    }

    if (p == 0) {   // one lane per head: packed f16 transposed output
        _Float16* wout = wgtT + ((size_t)bb * HW_ + hw) * INNER_ + head * DHEAD;
        #pragma unroll
        for (int j = 0; j < 4; ++j) {
            v8h pk;
            #pragma unroll
            for (int i = 0; i < 8; ++i) pk[i] = (_Float16)vacc[j * 8 + i];
            ((v8h*)wout)[j] = pk;
        }
    }
}

// =====================================================================
// host-side launcher
// =====================================================================
extern "C" void kernel_launch(void* const* d_in, const int* in_sizes, int n_in,
                              void* d_out, int out_size, void* d_ws, size_t ws_size,
                              hipStream_t stream) {
    const float* bev    = (const float*)d_in[0];
    const float* img    = (const float*)d_in[1];
    const float* Kmat   = (const float*)d_in[2];
    const float* Emat   = (const float*)d_in[3];
    const float* wxy    = (const float*)d_in[4];
    const float* wq     = (const float*)d_in[5];
    const float* bq     = (const float*)d_in[6];
    const float* wkv    = (const float*)d_in[7];
    const float* bkv    = (const float*)d_in[8];
    const float* w_off1 = (const float*)d_in[9];
    const float* b_off1 = (const float*)d_in[10];
    const float* w_off2 = (const float*)d_in[11];
    const float* b_off2 = (const float*)d_in[12];
    const float* w_proj = (const float*)d_in[13];
    const float* b_proj = (const float*)d_in[14];
    float* out = (float*)d_out;

    // ---- workspace partition ----
    char* ws = (char*)d_ws;
    float*    grid  = (float*)ws;     ws += (size_t)B_ * NCAM * HW_ * 2 * sizeof(float);
    float*    offb  = (float*)ws;     ws += (size_t)B_ * (2 * NPTS) * HW_ * sizeof(float);
    _Float16* qTb   = (_Float16*)ws;  ws += (size_t)B_ * HW_ * INNER_ * sizeof(_Float16);
    _Float16* o1T   = (_Float16*)ws;  ws += (size_t)B_ * HW_ * DIM_ * sizeof(_Float16);
    _Float16* wgtT  = (_Float16*)ws;  ws += (size_t)B_ * HW_ * INNER_ * sizeof(_Float16);
    _Float16* bevT  = (_Float16*)ws;  ws += (size_t)B_ * HW_ * DIM_ * sizeof(_Float16);
    _Float16* imgT  = (_Float16*)ws;  ws += (size_t)B_ * NCAM * HWIMG * DIM_ * sizeof(_Float16);
    _Float16* kvTb  = (_Float16*)ws;  ws += (size_t)B_ * NCAM * HWIMG * (2 * INNER_) * sizeof(_Float16);
    _Float16* wqH   = (_Float16*)ws;  ws += 16384 * sizeof(_Float16);
    _Float16* wkvH  = (_Float16*)ws;  ws += 32768 * sizeof(_Float16);
    _Float16* wo1H  = (_Float16*)ws;  ws += 16384 * sizeof(_Float16);
    _Float16* wo2H  = (_Float16*)ws;  ws += 2048  * sizeof(_Float16);
    _Float16* wprH  = (_Float16*)ws;

    // ---- 0) convert weights to f16 ----
    cvd_cvt_f16_kernel<<<64, 256, 0, stream>>>(wq,     wqH,  16384);
    cvd_cvt_f16_kernel<<<128, 256, 0, stream>>>(wkv,   wkvH, 32768);
    cvd_cvt_f16_kernel<<<64, 256, 0, stream>>>(w_off1, wo1H, 16384);
    cvd_cvt_f16_kernel<<<8,  256, 0, stream>>>(w_off2, wo2H, 2048);
    cvd_cvt_f16_kernel<<<64, 256, 0, stream>>>(w_proj, wprH, 16384);

    // ---- 0b) convert+transpose activations: [128,N] f32 -> [N,128] f16 ----
    {
        dim3 blk(16, 16);
        cvd_cvt_transpose_kernel<<<dim3(HW_ / 16, DIM_ / 16, B_), blk, 0, stream>>>(
            bev, bevT, DIM_, HW_);
        cvd_cvt_transpose_kernel<<<dim3(HWIMG / 16, DIM_ / 16, B_ * NCAM), blk, 0, stream>>>(
            img, imgT, DIM_, HWIMG);
    }

    // ---- 1) camera projection grid ----
    cvd_grid_kernel<<<(B_ * NCAM * HW_ + 255) / 256, 256, 0, stream>>>(
        Kmat, Emat, wxy, grid);

    // ---- 2) kvT[bn][hwimg][256] = (wkv @ img + bkv)^T, f16 ----
    cvd_wmma_gemm_kernel<<<512, 256, 0, stream>>>(
        wkvH, imgT, bkv, (void*)kvTb, 2 * INNER_, HWIMG, B_ * NCAM,
        (long)HWIMG * DIM_, (long)HWIMG * (2 * INNER_), 0, 1);

    // ---- 3) qT[b][hw][128] = (wq @ bev + bq)^T, f16 ----
    cvd_wmma_gemm_kernel<<<256, 256, 0, stream>>>(
        wqH, bevT, bq, (void*)qTb, INNER_, HW_, B_,
        (long)HW_ * DIM_, (long)HW_ * INNER_, 0, 1);

    // ---- 4) o1T[b][hw][128] = relu(w_off1 @ bev + b_off1)^T, f16 ----
    cvd_wmma_gemm_kernel<<<256, 256, 0, stream>>>(
        wo1H, bevT, b_off1, (void*)o1T, DIM_, HW_, B_,
        (long)HW_ * DIM_, (long)HW_ * DIM_, 1, 1);

    // ---- 5) off[b][16][hw] = w_off2 @ o1 + b_off2, f32 ----
    cvd_wmma_gemm_kernel<<<64, 256, 0, stream>>>(
        wo2H, o1T, b_off2, (void*)offb, 2 * NPTS, HW_, B_,
        (long)HW_ * DIM_, (long)(2 * NPTS) * HW_, 0, 0);

    // ---- 6) fused deformable sampling + attention -> wgtT f16 ----
    cvd_sample_attn_kernel<<<(B_ * HW_) / 8, 256, 0, stream>>>(
        grid, offb, qTb, kvTb, wgtT);

    // ---- 7) out[b][128][hw] = w_proj @ weighted + b_proj, f32 -> d_out ----
    cvd_wmma_gemm_kernel<<<256, 256, 0, stream>>>(
        wprH, wgtT, b_proj, (void*)out, DIM_, HW_, B_,
        (long)HW_ * INNER_, (long)DIM_ * HW_, 0, 0);
}
